// AttentionAugmentedAgent_16853451670059
// MI455X (gfx1250) — compile-verified, matrix operand order checked by reading, not run on hardware
//
#include <hip/hip_runtime.h>
#include <hip/hip_bf16.h>

typedef __attribute__((ext_vector_type(16))) _Float16 v16h;
typedef __attribute__((ext_vector_type(8)))  _Float16 v8h;
typedef __attribute__((ext_vector_type(8)))  float    v8f;

#define PI_F 3.14159265358979323846f

__device__ __forceinline__ float sigmoidf_(float x){ return 1.0f/(1.0f+expf(-x)); }

// Load one lane's half-row A/B fragment for v_wmma_f32_16x16x32_f16:
// elems 0..7 = K(k0+8h .. +7), elems 8..15 = K(k0+16+8h .. +7)  (two b128 loads)
__device__ __forceinline__ v16h ldfrag(const _Float16* __restrict__ row, int k0, int h){
  v8h x0 = *(const v8h*)(row + k0 + 8*h);
  v8h x1 = *(const v8h*)(row + k0 + 16 + 8*h);
  return __builtin_shufflevector(x0, x1, 0,1,2,3,4,5,6,7,8,9,10,11,12,13,14,15);
}

// ---------------------------------------------------------------- setup kernels
__global__ void spatial_basis_k(float* __restrict__ S){
  int idx = blockIdx.x*blockDim.x+threadIdx.x;
  if (idx >= 540*64) return;
  int c = idx & 63, p = idx >> 6;
  int y = p/20, x = p%20;
  int u = c >> 3, v = c & 7;
  S[idx] = cosf((float)((y+1)*(u+1))*(PI_F/27.0f)) *
           cosf((float)((x+1)*(v+1))*(PI_F/20.0f));
}

// f32 (R x Creal) -> f16 (R x Cpad) with zero pad (also used as N x K weight pack)
__global__ void convert_pad(const float* __restrict__ src, _Float16* __restrict__ dst,
                            int R, int Creal, int Cpad){
  int idx = blockIdx.x*blockDim.x+threadIdx.x;
  if (idx >= R*Cpad) return;
  int c = idx % Cpad, r = idx / Cpad;
  dst[idx] = (_Float16)(c < Creal ? src[(size_t)r*Creal + c] : 0.0f);
}

__global__ void pack_wg(const float* __restrict__ wxi,const float* __restrict__ whi,
                        const float* __restrict__ wxf,const float* __restrict__ whf,
                        const float* __restrict__ wxc,const float* __restrict__ whc,
                        const float* __restrict__ wxo,const float* __restrict__ who,
                        _Float16* __restrict__ dst){
  int idx = blockIdx.x*blockDim.x+threadIdx.x;
  if (idx >= 512*1728) return;
  int k = idx % 1728, n = idx / 1728;
  int g = n >> 7, oc = n & 127;
  const float* wx = (g==0)?wxi:(g==1)?wxf:(g==2)?wxc:wxo;
  const float* wh = (g==0)?whi:(g==1)?whf:(g==2)?whc:who;
  float v = (k < 576) ? wx[(size_t)oc*576 + k] : wh[(size_t)oc*1152 + (k-576)];
  dst[idx] = (_Float16)v;
}

__global__ void pack_bias512(const float* __restrict__ bxi,const float* __restrict__ bxf,
                             const float* __restrict__ bxc,const float* __restrict__ bxo,
                             float* __restrict__ dst){
  int n = blockIdx.x*blockDim.x+threadIdx.x;
  if (n >= 512) return;
  int g = n >> 7, oc = n & 127;
  dst[n] = (g==0)?bxi[oc]:(g==1)?bxf[oc]:(g==2)?bxc[oc]:bxo[oc];
}

__global__ void pack_lstm_w(const float* __restrict__ lwih,const float* __restrict__ lwhh,
                            _Float16* __restrict__ dst){
  int idx = blockIdx.x*blockDim.x+threadIdx.x;
  if (idx >= 1024*512) return;
  int k = idx % 512, n = idx / 512;
  float v = (k < 256) ? lwih[(size_t)n*256 + k] : lwhh[(size_t)n*256 + (k-256)];
  dst[idx] = (_Float16)v;
}

__global__ void pack_lstm_b(const float* __restrict__ lbih,const float* __restrict__ lbhh,
                            float* __restrict__ dst){
  int n = blockIdx.x*blockDim.x+threadIdx.x;
  if (n < 1024) dst[n] = lbih[n] + lbhh[n];
}

__global__ void pack_wpv(const float* __restrict__ pw,const float* __restrict__ vw,
                         _Float16* __restrict__ dst){
  int idx = blockIdx.x*blockDim.x+threadIdx.x;
  if (idx >= 32*256) return;
  int k = idx % 256, n = idx / 256;
  float v = (n < 18) ? pw[(size_t)n*256 + k] : (n == 18 ? vw[k] : 0.0f);
  dst[idx] = (_Float16)v;
}

__global__ void pack_bpv(const float* __restrict__ pb,const float* __restrict__ vb,
                         float* __restrict__ dst){
  int n = threadIdx.x;
  if (n < 19) dst[n] = (n < 18) ? pb[n] : vb[0];
}

// ---------------------------------------------------------------- conv1 (direct)
__global__ void conv1_k(const int* __restrict__ frame, const float* __restrict__ w,
                        const float* __restrict__ bias, float* __restrict__ z1, int t){
  int idx = blockIdx.x*blockDim.x+threadIdx.x;
  if (idx >= 16*32*52*39) return;
  int ox = idx % 39, oy = (idx/39) % 52, oc = (idx/(39*52)) % 32, b = idx/(39*52*32);
  const int* fb = frame + ((size_t)(t*16 + b))*3*210*160;
  float s = bias[oc];
  for (int c = 0; c < 3; ++c){
    const float* wr = w + (size_t)(oc*3 + c)*64;
    const int* fc = fb + (size_t)c*33600;
    for (int ky = 0; ky < 8; ++ky){
      int iy = oy*4 - 1 + ky;
      if ((unsigned)iy >= 210u) continue;
      for (int kx = 0; kx < 8; ++kx){
        int ix = ox*4 - 1 + kx;
        if ((unsigned)ix >= 160u) continue;
        s += (float)fc[iy*160 + ix] * (1.0f/255.0f) * wr[ky*8 + kx];
      }
    }
  }
  z1[idx] = s;
}

// ---------------------------------------------------------------- im2col (f32 -> f16 cols)
__global__ void im2col_k(const float* __restrict__ src, _Float16* __restrict__ dst,
                         const unsigned char* __restrict__ done, int t,
                         int B, int C, int H, int W, int OH, int OW, int KH, int KW,
                         int stride, int pad, int batchStride, int chanStride, int pixStride,
                         int colOfs, int Kpad, int useND){
  int total = B*OH*OW*C*KH*KW;
  int idx = blockIdx.x*blockDim.x+threadIdx.x;
  if (idx >= total) return;
  int kk  = idx % (KH*KW);
  int c   = (idx/(KH*KW)) % C;
  int pix = (idx/(KH*KW*C)) % (OH*OW);
  int b   = idx/(KH*KW*C*OH*OW);
  int ky = kk/KW, kx = kk%KW;
  int oy = pix/OW, ox = pix%OW;
  int iy = oy*stride - pad + ky, ix = ox*stride - pad + kx;
  float v = 0.0f;
  if (iy >= 0 && iy < H && ix >= 0 && ix < W){
    v = src[(size_t)b*batchStride + (size_t)c*chanStride + ((size_t)iy*W + ix)*(size_t)pixStride];
    if (useND) v *= (1.0f - (float)done[t*B + b]);
  }
  dst[((size_t)(b*OH*OW + pix))*Kpad + colOfs + (c*KH + ky)*KW + kx] = (_Float16)v;
}

// ---------------------------------------------------------------- WMMA GEMM (generic 1x1 tile)
// C[m,n] = sum_k A[m,k]*Bw[n,k] + bias[n]  (Bw stored N x K = transposed weights)
__global__ void gemm_wmma(const _Float16* __restrict__ A,
                          const _Float16* __restrict__ Bw,
                          const float*    __restrict__ bias,
                          float*          __restrict__ C,
                          _Float16*       __restrict__ C16,
                          int Mreal, int Nreal, int K, int ldc16, int relu){
  int lane = threadIdx.x;
  int r = lane & 15, h = lane >> 4;
  const _Float16* ar = A  + ((size_t)(blockIdx.x*16 + r))*K;
  const _Float16* br = Bw + ((size_t)(blockIdx.y*16 + r))*K;
  v8f acc = {};
  for (int k0 = 0; k0 < K; k0 += 32){
    v16h av = ldfrag(ar, k0, h);
    v16h bv = ldfrag(br, k0, h);
    acc = __builtin_amdgcn_wmma_f32_16x16x32_f16(false, av, false, bv, (short)0, acc, false, false);
  }
  int n = blockIdx.y*16 + r;
  float bval = (bias != nullptr && n < Nreal) ? bias[n] : 0.0f;
#pragma unroll
  for (int i = 0; i < 8; ++i){
    int m = blockIdx.x*16 + i + 8*h;   // C/D layout: VGPR i = row i + 8*half, lane&15 = col
    if (m >= Mreal) continue;
    float v = acc[i] + bval;
    if (relu) v = fmaxf(v, 0.0f);
    if (n < Nreal){
      if (C)   C  [(size_t)m*Nreal + n] = v;
      if (C16) C16[(size_t)m*ldc16 + n] = (_Float16)v;
    } else if (C16 && n < ldc16){
      C16[(size_t)m*ldc16 + n] = (_Float16)0.0f;
    }
  }
}

// ---------------------------------------------------------------- WMMA GEMM (2M x 4N register-blocked)
// Requires M % 32 == 0 and N % 64 == 0 (launch grid = (M/32, N/64)); N == ld of C.
// 8 WMMAs per K-step off 12 b128 loads: ~2.7x arithmetic intensity of the 1x1 version.
__global__ void gemm_wmma_2x4(const _Float16* __restrict__ A,
                              const _Float16* __restrict__ Bw,
                              const float*    __restrict__ bias,
                              float*          __restrict__ C,
                              int N, int K){
  int lane = threadIdx.x;
  int r = lane & 15, h = lane >> 4;
  const _Float16* ar0 = A + ((size_t)(blockIdx.x*32 + r))*K;
  const _Float16* ar1 = ar0 + (size_t)16*K;
  const _Float16* br0 = Bw + ((size_t)(blockIdx.y*64 + r))*K;
  const _Float16* br1 = br0 + (size_t)16*K;
  const _Float16* br2 = br0 + (size_t)32*K;
  const _Float16* br3 = br0 + (size_t)48*K;
  v8f acc00 = {}, acc01 = {}, acc02 = {}, acc03 = {};
  v8f acc10 = {}, acc11 = {}, acc12 = {}, acc13 = {};
  for (int k0 = 0; k0 < K; k0 += 32){
    v16h a0 = ldfrag(ar0, k0, h);
    v16h a1 = ldfrag(ar1, k0, h);
    v16h b0 = ldfrag(br0, k0, h);
    v16h b1 = ldfrag(br1, k0, h);
    v16h b2 = ldfrag(br2, k0, h);
    v16h b3 = ldfrag(br3, k0, h);
    acc00 = __builtin_amdgcn_wmma_f32_16x16x32_f16(false, a0, false, b0, (short)0, acc00, false, false);
    acc01 = __builtin_amdgcn_wmma_f32_16x16x32_f16(false, a0, false, b1, (short)0, acc01, false, false);
    acc02 = __builtin_amdgcn_wmma_f32_16x16x32_f16(false, a0, false, b2, (short)0, acc02, false, false);
    acc03 = __builtin_amdgcn_wmma_f32_16x16x32_f16(false, a0, false, b3, (short)0, acc03, false, false);
    acc10 = __builtin_amdgcn_wmma_f32_16x16x32_f16(false, a1, false, b0, (short)0, acc10, false, false);
    acc11 = __builtin_amdgcn_wmma_f32_16x16x32_f16(false, a1, false, b1, (short)0, acc11, false, false);
    acc12 = __builtin_amdgcn_wmma_f32_16x16x32_f16(false, a1, false, b2, (short)0, acc12, false, false);
    acc13 = __builtin_amdgcn_wmma_f32_16x16x32_f16(false, a1, false, b3, (short)0, acc13, false, false);
  }
  v8f* accs[2][4] = {{&acc00,&acc01,&acc02,&acc03},{&acc10,&acc11,&acc12,&acc13}};
#pragma unroll
  for (int j = 0; j < 4; ++j){
    int n = blockIdx.y*64 + j*16 + r;
    float bval = bias ? bias[n] : 0.0f;
#pragma unroll
    for (int i = 0; i < 2; ++i){
      v8f a = *accs[i][j];
#pragma unroll
      for (int v = 0; v < 8; ++v){
        int m = blockIdx.x*32 + i*16 + v + 8*h;
        C[(size_t)m*N + n] = a[v] + bval;
      }
    }
  }
}

// ---------------------------------------------------------------- ConvLSTM pointwise
__global__ void convlstm_pw(const float* __restrict__ gates,
                            const float* __restrict__ peepI, const float* __restrict__ peepF,
                            const float* __restrict__ peepO,
                            float* __restrict__ vh, float* __restrict__ vc,
                            const unsigned char* __restrict__ done, int t){
  int idx = blockIdx.x*blockDim.x+threadIdx.x;
  if (idx >= 16*128*540) return;
  int p  = idx % 540;
  int oc = (idx/540) & 127;
  int b  = idx/(540*128);
  float nd = 1.0f - (float)done[t*16 + b];
  float cp = vc[idx] * nd;
  size_t gbase = ((size_t)(b*540 + p))*512 + oc;
  float gi = gates[gbase      ];
  float gf = gates[gbase + 128];
  float gc = gates[gbase + 256];
  float go = gates[gbase + 384];
  float wi = peepI[oc*540 + p], wf = peepF[oc*540 + p], wo = peepO[oc*540 + p];
  float ci = sigmoidf_(gi + cp*wi);
  float cf = sigmoidf_(gf + cp*wf);
  float cc = cf*cp + ci*tanhf(gc);
  float co = sigmoidf_(go + cc*wo);
  vc[idx] = cc;
  vh[idx] = co * tanhf(cc);
}

// ---------------------------------------------------------------- attention
__global__ void attn_logits(const float* __restrict__ vh, const float* __restrict__ S,
                            const float* __restrict__ Q, float* __restrict__ logits){
  int idx = blockIdx.x*blockDim.x+threadIdx.x;
  if (idx >= 16*4*540) return;
  int p = idx % 540, q = (idx/540) & 3, b = idx/(540*4);
  const float* Qr = Q + (size_t)b*288 + q*72;
  float s = 0.0f;
  for (int c = 0; c < 8; ++c)  s += vh[((size_t)(b*128 + c))*540 + p] * Qr[c];
  const float* Sp = S + (size_t)p*64;
  for (int c = 0; c < 64; ++c) s += Sp[c] * Qr[8 + c];
  logits[(size_t)(b*4 + q)*540 + p] = s;
}

__global__ void softmax540(float* __restrict__ L){
  int row = blockIdx.x, lane = threadIdx.x;   // 32 lanes
  float* Lr = L + (size_t)row*540;
  float mx = -1e30f;
  for (int p = lane; p < 540; p += 32) mx = fmaxf(mx, Lr[p]);
  for (int s = 16; s; s >>= 1) mx = fmaxf(mx, __shfl_xor(mx, s, 32));
  float sum = 0.0f;
  for (int p = lane; p < 540; p += 32) sum += expf(Lr[p] - mx);
  for (int s = 16; s; s >>= 1) sum += __shfl_xor(sum, s, 32);
  float inv = 1.0f/sum;
  for (int p = lane; p < 540; p += 32) Lr[p] = expf(Lr[p] - mx) * inv;
}

__global__ void attn_ans(const float* __restrict__ A, const float* __restrict__ vh,
                         const float* __restrict__ S, float* __restrict__ ans){
  int idx = blockIdx.x*blockDim.x+threadIdx.x;
  if (idx >= 16*4*184) return;
  int d = idx % 184, q = (idx/184) & 3, b = idx/(184*4);
  const float* Ar = A + (size_t)(b*4 + q)*540;
  float s = 0.0f;
  if (d < 120){
    const float* Vr = vh + ((size_t)(b*128 + 8 + d))*540;
    for (int p = 0; p < 540; ++p) s += Ar[p] * Vr[p];
  } else {
    int c = d - 120;
    for (int p = 0; p < 540; ++p) s += Ar[p] * S[(size_t)p*64 + c];
  }
  ans[((size_t)b*4 + q)*184 + d] = s;
}

__global__ void feat_assemble(const float* __restrict__ ans, const float* __restrict__ Q,
                              const float* __restrict__ reward, const int* __restrict__ la,
                              int t, float* __restrict__ feat){
  int idx = blockIdx.x*blockDim.x+threadIdx.x;
  if (idx >= 16*1043) return;
  int j = idx % 1043, b = idx / 1043;
  float v;
  if      (j < 736)   v = ans[(size_t)b*736 + j];
  else if (j < 1024)  v = Q[(size_t)b*288 + (j - 736)];
  else if (j == 1024) v = reward[t*16 + b];
  else                v = (la[t*16 + b] == (j - 1025)) ? 1.0f : 0.0f;
  feat[(size_t)b*1043 + j] = v;
}

// ---------------------------------------------------------------- LSTM
__global__ void lstm_prepA(const float* __restrict__ core, const float* __restrict__ ph,
                           const unsigned char* __restrict__ done, int t,
                           _Float16* __restrict__ Aout){
  int idx = blockIdx.x*blockDim.x+threadIdx.x;
  if (idx >= 16*512) return;
  int k = idx & 511, b = idx >> 9;
  float v = (k < 256) ? core[b*256 + k]
                      : ph[b*256 + (k - 256)] * (1.0f - (float)done[t*16 + b]);
  Aout[idx] = (_Float16)v;
}

__global__ void lstm_pw(const float* __restrict__ lg, float* __restrict__ pc,
                        float* __restrict__ ph, const unsigned char* __restrict__ done,
                        int t, _Float16* __restrict__ outs16){
  int idx = blockIdx.x*blockDim.x+threadIdx.x;
  if (idx >= 16*256) return;
  int j = idx & 255, b = idx >> 8;
  float nd = 1.0f - (float)done[t*16 + b];
  const float* g = lg + (size_t)b*1024;
  float pcg = pc[idx] * nd;
  float pcn = sigmoidf_(g[256 + j])*pcg + sigmoidf_(g[j])*tanhf(g[512 + j]);
  float phn = sigmoidf_(g[768 + j])*tanhf(pcn);
  pc[idx] = pcn;
  ph[idx] = phn;
  outs16[((size_t)(t*16 + b))*256 + j] = (_Float16)phn;
}

// ---------------------------------------------------------------- host
static inline int cdiv(int a, int b){ return (a + b - 1)/b; }

extern "C" void kernel_launch(void* const* d_in, const int* in_sizes, int n_in,
                              void* d_out, int out_size, void* d_ws, size_t ws_size,
                              hipStream_t stream){
  (void)in_sizes; (void)n_in; (void)out_size;
  const int T = 4;

  const int*   frame       = (const int*)d_in[0];
  const float* reward      = (const float*)d_in[1];
  const int*   last_action = (const int*)d_in[2];
  const unsigned char* done= (const unsigned char*)d_in[3];
  const float* state_vh    = (const float*)d_in[4];
  const float* state_vc    = (const float*)d_in[5];
  const float* state_ph    = (const float*)d_in[6];
  const float* state_pc    = (const float*)d_in[7];
  const float* c1w=(const float*)d_in[8],  *c1b=(const float*)d_in[9];
  const float* c2w=(const float*)d_in[10], *c2b=(const float*)d_in[11];
  const float* Wxi=(const float*)d_in[12], *bxi=(const float*)d_in[13], *Whi=(const float*)d_in[14];
  const float* Wxf=(const float*)d_in[15], *bxf=(const float*)d_in[16], *Whf=(const float*)d_in[17];
  const float* Wxc=(const float*)d_in[18], *bxc=(const float*)d_in[19], *Whc=(const float*)d_in[20];
  const float* Wxo=(const float*)d_in[21], *bxo=(const float*)d_in[22], *Who=(const float*)d_in[23];
  const float* Wci=(const float*)d_in[24], *Wcf=(const float*)d_in[25], *Wco=(const float*)d_in[26];
  const float* q1w=(const float*)d_in[27], *q1b=(const float*)d_in[28];
  const float* q2w=(const float*)d_in[29], *q2b=(const float*)d_in[30];
  const float* q3w=(const float*)d_in[31], *q3b=(const float*)d_in[32];
  const float* a1w=(const float*)d_in[33], *a1b=(const float*)d_in[34];
  const float* a2w=(const float*)d_in[35], *a2b=(const float*)d_in[36];
  const float* lwih=(const float*)d_in[37], *lwhh=(const float*)d_in[38];
  const float* lbih=(const float*)d_in[39], *lbhh=(const float*)d_in[40];
  const float* pw=(const float*)d_in[41],  *pb=(const float*)d_in[42];
  const float* vw=(const float*)d_in[43],  *vb=(const float*)d_in[44];

  size_t off = 0;
  auto alloc = [&](size_t bytes)->char*{
    off = (off + 255) & ~(size_t)255;
    char* p = (char*)d_ws + off;
    off += bytes;
    return p;
  };
  float*    Sb    = (float*)    alloc((size_t)540*64*4);
  _Float16* W2p   = (_Float16*) alloc((size_t)64*512*2);
  _Float16* Wgp   = (_Float16*) alloc((size_t)512*1728*2);
  float*    b512  = (float*)    alloc(512*4);
  _Float16* Wq1p  = (_Float16*) alloc((size_t)128*256*2);
  _Float16* Wq2p  = (_Float16*) alloc((size_t)288*128*2);
  _Float16* Wq3p  = (_Float16*) alloc((size_t)288*288*2);
  _Float16* Wa1p  = (_Float16*) alloc((size_t)512*1056*2);
  _Float16* Wa2p  = (_Float16*) alloc((size_t)256*512*2);
  _Float16* Wlp   = (_Float16*) alloc((size_t)1024*512*2);
  float*    blp   = (float*)    alloc(1024*4);
  _Float16* Wpvp  = (_Float16*) alloc((size_t)32*256*2);
  float*    bpv   = (float*)    alloc(19*4);
  float*    vhb   = (float*)    alloc((size_t)16*128*540*4);
  float*    vcb   = (float*)    alloc((size_t)16*128*540*4);
  float*    phb   = (float*)    alloc((size_t)16*256*4);
  float*    pcb   = (float*)    alloc((size_t)16*256*4);
  float*    z1    = (float*)    alloc((size_t)16*32*52*39*4);
  _Float16* A2    = (_Float16*) alloc((size_t)8640*512*2);
  float*    z2m   = (float*)    alloc((size_t)8640*64*4);
  _Float16* Acol  = (_Float16*) alloc((size_t)8640*1728*2);
  float*    gatesb= (float*)    alloc((size_t)8640*512*4);
  _Float16* ph16  = (_Float16*) alloc((size_t)16*256*2);
  _Float16* q116  = (_Float16*) alloc((size_t)16*128*2);
  _Float16* q216  = (_Float16*) alloc((size_t)16*288*2);
  float*    Qb    = (float*)    alloc((size_t)16*288*4);
  float*    logb  = (float*)    alloc((size_t)64*540*4);
  float*    ansb  = (float*)    alloc((size_t)16*736*4);
  float*    featb = (float*)    alloc((size_t)16*1043*4);
  _Float16* feat16= (_Float16*) alloc((size_t)16*1056*2);
  _Float16* a116  = (_Float16*) alloc((size_t)16*512*2);
  float*    coreb = (float*)    alloc((size_t)16*256*4);
  _Float16* lA    = (_Float16*) alloc((size_t)16*512*2);
  float*    lg    = (float*)    alloc((size_t)16*1024*4);
  _Float16* outs16= (_Float16*) alloc((size_t)64*256*2);
  if (off > ws_size) return;

  // ---- per-call setup: spatial basis, weight packs (f16, N x K), state init
  spatial_basis_k<<<cdiv(540*64,256),256,0,stream>>>(Sb);
  convert_pad<<<cdiv(64*512,256),256,0,stream>>>(c2w, W2p, 64, 512, 512);
  pack_wg<<<cdiv(512*1728,256),256,0,stream>>>(Wxi,Whi,Wxf,Whf,Wxc,Whc,Wxo,Who,Wgp);
  pack_bias512<<<2,256,0,stream>>>(bxi,bxf,bxc,bxo,b512);
  convert_pad<<<cdiv(128*256,256),256,0,stream>>>(q1w, Wq1p, 128, 256, 256);
  convert_pad<<<cdiv(288*128,256),256,0,stream>>>(q2w, Wq2p, 288, 128, 128);
  convert_pad<<<cdiv(288*288,256),256,0,stream>>>(q3w, Wq3p, 288, 288, 288);
  convert_pad<<<cdiv(512*1056,256),256,0,stream>>>(a1w, Wa1p, 512, 1043, 1056);
  convert_pad<<<cdiv(256*512,256),256,0,stream>>>(a2w, Wa2p, 256, 512, 512);
  pack_lstm_w<<<cdiv(1024*512,256),256,0,stream>>>(lwih, lwhh, Wlp);
  pack_lstm_b<<<4,256,0,stream>>>(lbih, lbhh, blp);
  pack_wpv<<<cdiv(32*256,256),256,0,stream>>>(pw, vw, Wpvp);
  pack_bpv<<<1,32,0,stream>>>(pb, vb, bpv);
  hipMemcpyAsync(vhb, state_vh, (size_t)16*128*540*4, hipMemcpyDeviceToDevice, stream);
  hipMemcpyAsync(vcb, state_vc, (size_t)16*128*540*4, hipMemcpyDeviceToDevice, stream);
  hipMemcpyAsync(phb, state_ph, (size_t)16*256*4,     hipMemcpyDeviceToDevice, stream);
  hipMemcpyAsync(pcb, state_pc, (size_t)16*256*4,     hipMemcpyDeviceToDevice, stream);

  for (int t = 0; t < T; ++t){
    // conv1 direct: (16,3,210,160) -> (16,32,52,39)
    conv1_k<<<cdiv(16*32*52*39,256),256,0,stream>>>(frame, c1w, c1b, z1, t);
    // conv2 as implicit GEMM: im2col (8640 x 512) then blocked WMMA (M=270*32, N=1*64)
    im2col_k<<<cdiv(16*540*32*16,256),256,0,stream>>>(z1, A2, done, t,
        16,32,52,39, 27,20, 4,4, 2,2, 32*52*39, 52*39, 1, 0, 512, 0);
    gemm_wmma_2x4<<<dim3(270,1),32,0,stream>>>(A2, W2p, c2b, z2m, 64, 512);
    // ConvLSTM fused-gate implicit GEMM: cols [0,576) = z (HWC), [576,1728) = h*nd (CHW)
    im2col_k<<<cdiv(16*540*64*9,256),256,0,stream>>>(z2m, Acol, done, t,
        16,64,27,20, 27,20, 3,3, 1,1, 540*64, 1, 64, 0, 1728, 0);
    im2col_k<<<cdiv(16*540*128*9,256),256,0,stream>>>(vhb, Acol, done, t,
        16,128,27,20, 27,20, 3,3, 1,1, 128*540, 540, 1, 576, 1728, 1);
    gemm_wmma_2x4<<<dim3(270,8),32,0,stream>>>(Acol, Wgp, b512, gatesb, 512, 1728);
    convlstm_pw<<<cdiv(16*128*540,256),256,0,stream>>>(gatesb, Wci, Wcf, Wco, vhb, vcb, done, t);
    // query MLP from ph (pre-update)
    convert_pad<<<cdiv(16*256,256),256,0,stream>>>(phb, ph16, 16, 256, 256);
    gemm_wmma<<<dim3(1,8),32,0,stream>>>(ph16, Wq1p, q1b, nullptr, q116, 16, 128, 256, 128, 1);
    gemm_wmma<<<dim3(1,18),32,0,stream>>>(q116, Wq2p, q2b, nullptr, q216, 16, 288, 128, 288, 1);
    gemm_wmma<<<dim3(1,18),32,0,stream>>>(q216, Wq3p, q3b, Qb, nullptr, 16, 288, 288, 0, 0);
    // attention over 540 positions
    attn_logits<<<cdiv(16*4*540,256),256,0,stream>>>(vhb, Sb, Qb, logb);
    softmax540<<<64,32,0,stream>>>(logb);
    attn_ans<<<cdiv(16*4*184,256),256,0,stream>>>(logb, vhb, Sb, ansb);
    feat_assemble<<<cdiv(16*1043,256),256,0,stream>>>(ansb, Qb, reward, last_action, t, featb);
    convert_pad<<<cdiv(16*1056,256),256,0,stream>>>(featb, feat16, 16, 1043, 1056);
    gemm_wmma<<<dim3(1,32),32,0,stream>>>(feat16, Wa1p, a1b, nullptr, a116, 16, 512, 1056, 512, 1);
    gemm_wmma<<<dim3(1,16),32,0,stream>>>(a116, Wa2p, a2b, coreb, nullptr, 16, 256, 512, 0, 0);
    // LSTM core
    lstm_prepA<<<cdiv(16*512,256),256,0,stream>>>(coreb, phb, done, t, lA);
    gemm_wmma<<<dim3(1,64),32,0,stream>>>(lA, Wlp, blp, lg, nullptr, 16, 1024, 512, 0, 0);
    lstm_pw<<<cdiv(16*256,256),256,0,stream>>>(lg, pcb, phb, done, t, outs16);
  }
  // policy/value heads: (64 x 256) @ (256 x 19)
  gemm_wmma<<<dim3(4,2),32,0,stream>>>(outs16, Wpvp, bpv, (float*)d_out, nullptr, 64, 19, 256, 0, 0);
}
